// DeformConv2d_30880814858420
// MI455X (gfx1250) — matrix operand order repaired and checked by reading, not compiled, not run on hardware
//
#include <hip/hip_runtime.h>

// ---------------------------------------------------------------------------
// DeformConv2d forward for MI455X (gfx1250, wave32).
//   B=8, Cin=128, H=W=112, Cout=256, K=3, pad=1, stride=1.
// Hot stage: implicit GEMM (M=100352, N=256, K=1152, 59 GFLOP) on
// v_wmma_f32_16x16x32_bf16.  A and B tiles both staged with
// GLOBAL_LOAD_ASYNC_TO_LDS_B128 in a 3-deep ASYNCcnt-counted pipeline.
// Sampled map stored with a 1-px zero halo so A staging needs no bounds check.
// ---------------------------------------------------------------------------

#define Wd     112
#define HW     12544          // 112*112
#define HWD    114            // haloed width
#define CIN    128
#define COUT   256
#define NBAT   8
#define NKSTEP 36             // K = 9*128 split into 36 steps of 32
#define LDSP   40             // padded LDS row stride (elements): 80 B rows
#define MTILE  64             // pixels per workgroup
#define NBUF   3              // async pipeline depth

#if __has_builtin(__builtin_amdgcn_global_load_async_to_lds_b128) && \
    __has_builtin(__builtin_amdgcn_s_wait_asynccnt)
#define HAVE_ASYNC 1
#else
#define HAVE_ASYNC 0
#endif

#define GLOBAL_AS __attribute__((address_space(1)))
#define LDS_AS    __attribute__((address_space(3)))

typedef int v4i __attribute__((vector_size(16)));   // matches builtin param type
typedef __attribute__((ext_vector_type(16))) __bf16 v16bf;
typedef __attribute__((ext_vector_type(8)))  float  v8f;

__device__ __forceinline__ unsigned short f2bf(float f) {
    union { float f; unsigned int u; } c; c.f = f;
    unsigned int u = c.u;
    u += 0x7fffu + ((u >> 16) & 1u);       // round-to-nearest-even
    return (unsigned short)(u >> 16);
}

// ---------------------------------------------------------------------------
// Kernel Z: zero the haloed sampled buffer (one b128 store per thread).
// ---------------------------------------------------------------------------
__global__ void k_zero(uint4* __restrict__ p) {
    size_t i = (size_t)blockIdx.x * 256 + threadIdx.x;
    uint4 z; z.x = z.y = z.z = z.w = 0u;
    p[i] = z;
}

// ---------------------------------------------------------------------------
// Kernel 0: repack conv_w (256,128,3,3) f32 -> bf16, layout [s][n][kk]:
//   s = r*4 + cb (r = ky*3+kx, cb = ci/32), kk = ci%32.
// ---------------------------------------------------------------------------
__global__ void k_pack_w(const float* __restrict__ w, unsigned short* __restrict__ wp) {
    int idx = blockIdx.x * 256 + threadIdx.x;        // 36*256*32 = 294912
    int kk = idx & 31;
    int n  = (idx >> 5) & 255;
    int s  = idx >> 13;
    int r  = s >> 2, cb = s & 3;
    int ci = cb * 32 + kk;
    wp[idx] = f2bf(w[(n * CIN + ci) * 9 + r]);
}

// ---------------------------------------------------------------------------
// Kernel 1: offset conv (Cout=18). One thread per pixel, 18 accumulators.
// ---------------------------------------------------------------------------
__global__ void k_offset(const float* __restrict__ x, const float* __restrict__ ow,
                         const float* __restrict__ ob, float* __restrict__ off) {
    int pix = blockIdx.x * 256 + threadIdx.x;        // 49 blocks * 256 = 12544
    int bb  = blockIdx.y;
    int y = pix / Wd, x0 = pix % Wd;
    float acc[18];
    #pragma unroll
    for (int o = 0; o < 18; ++o) acc[o] = ob[o];
    const float* xb = x + (size_t)bb * CIN * HW;
    for (int ci = 0; ci < CIN; ++ci) {
        const float* xc = xb + ci * HW;
        float xv[9];
        #pragma unroll
        for (int r = 0; r < 9; ++r) {
            int yy = y + r / 3 - 1, xx = x0 + r % 3 - 1;
            bool ok = (yy >= 0) && (yy < Wd) && (xx >= 0) && (xx < Wd);
            xv[r] = ok ? xc[yy * Wd + xx] : 0.0f;
        }
        #pragma unroll
        for (int o = 0; o < 18; ++o) {
            const float* wr = ow + (o * CIN + ci) * 9;
            #pragma unroll
            for (int r = 0; r < 9; ++r) acc[o] = fmaf(wr[r], xv[r], acc[o]);
        }
    }
    #pragma unroll
    for (int o = 0; o < 18; ++o) off[((size_t)bb * 18 + o) * HW + pix] = acc[o];
}

// ---------------------------------------------------------------------------
// Kernel 2: bilinear sample + sum over 9 points -> channels-last bf16 map
// with 1-px zero halo: s_cl[b][y+1][x+1][ci], dims (8,114,114,128).
// ---------------------------------------------------------------------------
__global__ void k_sample(const float* __restrict__ x, const float* __restrict__ off,
                         unsigned short* __restrict__ s_cl) {
    int bb = blockIdx.y;
    int t  = threadIdx.x;
    int px = t & 31, g = t >> 5;                     // g = ci-group (16 ch each)
    int pix = blockIdx.x * 32 + px;
    int y = pix / Wd, x0 = pix % Wd;
    float acc[16];
    #pragma unroll
    for (int c = 0; c < 16; ++c) acc[c] = 0.0f;
    const float* xb = x + ((size_t)bb * CIN + g * 16) * HW;
    const float* ofb = off + (size_t)bb * 18 * HW + pix;
    for (int r = 0; r < 9; ++r) {
        float sy = (float)(y + r / 3 - 1) + ofb[(size_t)r * HW];
        float sx = (float)(x0 + r % 3 - 1) + ofb[(size_t)(9 + r) * HW];
        float fy0 = floorf(sy), fx0 = floorf(sx);
        int   y0 = (int)fy0,   xq = (int)fx0;
        float wy1 = sy - fy0,  wx1 = sx - fx0;
        float wy0 = 1.0f - wy1, wx0 = 1.0f - wx1;
        #pragma unroll
        for (int cy = 0; cy < 2; ++cy) {
            int yy = y0 + cy;
            float wy = cy ? wy1 : wy0;
            bool vy = (yy >= 0) && (yy < Wd);
            int yc = min(max(yy, 0), Wd - 1);
            #pragma unroll
            for (int cx = 0; cx < 2; ++cx) {
                int xx = xq + cx;
                float wx = cx ? wx1 : wx0;
                bool vx = (xx >= 0) && (xx < Wd);
                int xc = min(max(xx, 0), Wd - 1);
                float wgt = (vy && vx) ? wy * wx : 0.0f;
                const float* p = xb + yc * Wd + xc;
                #pragma unroll
                for (int c = 0; c < 16; ++c)
                    acc[c] = fmaf(wgt, p[(size_t)c * HW], acc[c]);
            }
        }
    }
    union { unsigned short h[16]; uint4 q[2]; } u;
    #pragma unroll
    for (int c = 0; c < 16; ++c) u.h[c] = f2bf(acc[c]);
    unsigned short* dst = s_cl + (((size_t)bb * HWD + (y + 1)) * HWD + (x0 + 1)) * CIN + g * 16;
    *(uint4*)dst       = u.q[0];
    *(uint4*)(dst + 8) = u.q[1];
}

// ---------------------------------------------------------------------------
// Per-K-step compute: 2 A-frags x 4 B-frags -> 8 WMMAs per wave.
// A frag (ISA 16-bit A 16x32): lane<16 K in {0..7,16..23}; lane>=16 +8.
// B frag (ISA 16-bit B 32x16): lane<16 K in {0..7,16..23}; lane>=16 +8.
// ---------------------------------------------------------------------------
__device__ __forceinline__ void mma_step(const unsigned short* Ab, const unsigned short* Bb,
                                         int wm, int wn, int lane, v8f (&acc)[2][4]) {
    int ml = lane & 15, hi = lane >> 4;
    int abase = hi * 8;
    v16bf afrag[2];
    #pragma unroll
    for (int i2 = 0; i2 < 2; ++i2) {
        int m = wm * 32 + i2 * 16 + ml;
        uint4* ap = (uint4*)&afrag[i2];
        ap[0] = *(const uint4*)&Ab[m * LDSP + abase];
        ap[1] = *(const uint4*)&Ab[m * LDSP + 16 + abase];
    }
    int bbase = hi * 8;
    #pragma unroll
    for (int j = 0; j < 4; ++j) {
        int n = wn * 64 + j * 16 + ml;
        v16bf bfrag;
        uint4* bp = (uint4*)&bfrag;
        bp[0] = *(const uint4*)&Bb[n * LDSP + bbase];
        bp[1] = *(const uint4*)&Bb[n * LDSP + bbase + 16];
        #pragma unroll
        for (int i2 = 0; i2 < 2; ++i2)
            acc[i2][j] = __builtin_amdgcn_wmma_f32_16x16x32_bf16(
                             false, afrag[i2], false, bfrag, (short)0, acc[i2][j], false, false);
    }
}

// ---------------------------------------------------------------------------
// Kernel 3: implicit-GEMM 3x3 conv on WMMA bf16.
//   Workgroup: 256 thr (8 waves) -> 64 pixels x 256 Cout tile.
//   Fully-async tile staging, 3-deep pipeline, exactly 5 async instrs per
//   wave per K-step (4 B + 1 A) -> s_wait_asynccnt(5) retires step s while
//   step s+1 stays in flight; step s+2 issued post-barrier.
// ---------------------------------------------------------------------------
__global__ void __launch_bounds__(256)
k_conv(const unsigned short* __restrict__ s_cl, const unsigned short* __restrict__ wp,
       float* __restrict__ out) {
    __shared__ __align__(16) unsigned short Abuf[NBUF][MTILE * LDSP];   // 15 KB
    __shared__ __align__(16) unsigned short Bbuf[NBUF][256 * LDSP];     // 60 KB

    int bb = blockIdx.y;
    int pb = blockIdx.x * MTILE;
    int t  = threadIdx.x;
    int wave = t >> 5, lane = t & 31;
    int wm = wave & 1, wn = wave >> 1;               // 2(M) x 4(N) wave grid

    // A-staging role: thread -> (pixel, 8-channel chunk of 16 B)
    int a_px = t >> 2;                               // 0..63
    int a_cp = t & 3;
    int apix = pb + a_px;
    int ay = apix / Wd, ax = apix % Wd;

    v8f acc[2][4];
    #pragma unroll
    for (int i2 = 0; i2 < 2; ++i2)
        #pragma unroll
        for (int j = 0; j < 4; ++j) acc[i2][j] = (v8f){};

    const unsigned short* sb = s_cl + (size_t)bb * HWD * HWD * CIN;

#if HAVE_ASYNC
    auto aissue = [&](int s) {       // 1 async b128: haloed -> always in-bounds
        int r = s >> 2, cb = s & 3;
        int hrow = ay + r / 3, hcol = ax + r % 3;    // halo coords (dy+1, dx+1)
        GLOBAL_AS v4i* g = (GLOBAL_AS v4i*)(sb + ((size_t)hrow * HWD + hcol) * CIN
                                               + cb * 32 + a_cp * 8);
        LDS_AS v4i* l = (LDS_AS v4i*)&Abuf[s % NBUF][a_px * LDSP + a_cp * 8];
        __builtin_amdgcn_global_load_async_to_lds_b128(g, l, 0, 0);
    };
    auto bissue = [&](int s) {       // 4 async b128: thread t copies row n=t
        GLOBAL_AS v4i* g = (GLOBAL_AS v4i*)(wp + ((size_t)s * 256 + t) * 32);
        LDS_AS    v4i* l = (LDS_AS    v4i*)&Bbuf[s % NBUF][t * LDSP];
        __builtin_amdgcn_global_load_async_to_lds_b128(g, l, 0, 0);
        __builtin_amdgcn_global_load_async_to_lds_b128(g, l, 16, 0);
        __builtin_amdgcn_global_load_async_to_lds_b128(g, l, 32, 0);
        __builtin_amdgcn_global_load_async_to_lds_b128(g, l, 48, 0);
    };

    aissue(0); bissue(0);
    aissue(1); bissue(1);
    for (int s = 0; s < NKSTEP; ++s) {
        int buf = s % NBUF;
        if (s + 1 < NKSTEP) __builtin_amdgcn_s_wait_asynccnt(5);  // step s done
        else                __builtin_amdgcn_s_wait_asynccnt(0);
        __syncthreads();
        if (s + 2 < NKSTEP) { aissue(s + 2); bissue(s + 2); }     // buf free
        mma_step(Abuf[buf], Bbuf[buf], wm, wn, lane, acc);
    }
#else
    // ---- fallback: register-prefetch staging (proven path), haloed reads.
    auto aload = [&](int s) -> uint4 {
        int r = s >> 2, cb = s & 3;
        int hrow = ay + r / 3, hcol = ax + r % 3;
        return *(const uint4*)(sb + ((size_t)hrow * HWD + hcol) * CIN + cb * 32 + a_cp * 8);
    };
    uint4 aReg = aload(0);
    uint4 bReg[4];
    {
        const uint4* bp = (const uint4*)(wp + (size_t)t * 32);
        bReg[0] = bp[0]; bReg[1] = bp[1]; bReg[2] = bp[2]; bReg[3] = bp[3];
    }
    for (int s = 0; s < NKSTEP; ++s) {
        int buf = s % NBUF;
        *(uint4*)&Abuf[buf][a_px * LDSP + a_cp * 8] = aReg;
        {
            uint4* br = (uint4*)&Bbuf[buf][t * LDSP];
            br[0] = bReg[0]; br[1] = bReg[1]; br[2] = bReg[2]; br[3] = bReg[3];
        }
        if (s + 1 < NKSTEP) {
            aReg = aload(s + 1);
            const uint4* bp = (const uint4*)(wp + ((size_t)(s + 1) * 256 + t) * 32);
            bReg[0] = bp[0]; bReg[1] = bp[1]; bReg[2] = bp[2]; bReg[3] = bp[3];
        }
        __syncthreads();
        mma_step(Abuf[buf], Bbuf[buf], wm, wn, lane, acc);
    }
#endif

    // Epilogue. C/D layout: VGPR i -> M = i + 8*(lane>=16); N = lane&15.
    int hi = lane >> 4, ml = lane & 15;
    #pragma unroll
    for (int i2 = 0; i2 < 2; ++i2) {
        int p = pb + wm * 32 + i2 * 16 + hi * 8;
        #pragma unroll
        for (int j = 0; j < 4; ++j) {
            int co = wn * 64 + j * 16 + ml;
            float* orow = out + ((size_t)bb * COUT + co) * HW + p;
            union { v8f v; float4 q[2]; } u;
            u.v = acc[i2][j];
            *(float4*)orow       = u.q[0];
            *(float4*)(orow + 4) = u.q[1];
        }
    }
}

// ---------------------------------------------------------------------------
// Host launcher.  Workspace layout (all fully written before read):
//   ws_off : f32  [8][18][12544]              7.2 MB
//   ws_s   : bf16 [8][114][114][128]         26.6 MB  (haloed, channels-last)
//   ws_w   : bf16 [36][256][32]               0.6 MB  (packed weights)
// ---------------------------------------------------------------------------
extern "C" void kernel_launch(void* const* d_in, const int* in_sizes, int n_in,
                              void* d_out, int out_size, void* d_ws, size_t ws_size,
                              hipStream_t stream) {
    const float* x  = (const float*)d_in[0];
    const float* ow = (const float*)d_in[1];
    const float* ob = (const float*)d_in[2];
    const float* cw = (const float*)d_in[3];
    float* out = (float*)d_out;

    float*          ws_off = (float*)d_ws;
    unsigned short* ws_s   = (unsigned short*)(ws_off + (size_t)NBAT * 18 * HW);
    unsigned short* ws_w   = ws_s + (size_t)NBAT * HWD * HWD * CIN;

    // 8*114*114*128 bf16 = 1,663,488 uint4 -> exactly 6498 blocks of 256.
    k_zero  <<<6498, 256, 0, stream>>>((uint4*)ws_s);
    k_pack_w<<<1152, 256, 0, stream>>>(cw, ws_w);
    k_offset<<<dim3(49,  NBAT), 256, 0, stream>>>(x, ow, ob, ws_off);
    k_sample<<<dim3(392, NBAT), 256, 0, stream>>>(x, ws_off, ws_s);
    k_conv  <<<dim3(196, NBAT), 256, 0, stream>>>(ws_s, ws_w, out);
}
